// SpotMamba_15247133901395
// MI455X (gfx1250) — compile-verified, matrix operand order checked by reading, not run on hardware
//
#include <hip/hip_runtime.h>
#include <hip/hip_bf16.h>

// ---------------------------------------------------------------------------
// SpotMamba (Vision Mamba) forward for MI455X / gfx1250.
// GEMMs use v_wmma_f32_16x16x32_bf16 (wave32 WMMA, f32 accumulate).
// ---------------------------------------------------------------------------

typedef __attribute__((ext_vector_type(16))) __bf16 v16bf;
typedef __attribute__((ext_vector_type(8)))  float  v8f;

union FragBf { uint4 u[2]; v16bf v; };

#define NTOK   6272      // 2 batches * 3136 tokens
#define LSEQ   3136
#define DM     384
#define DI     768
#define DSTATE 14
#define DTRANK 24

// ------------------------- small helpers -----------------------------------

static __device__ __forceinline__ unsigned short f2bf(float f) {
  unsigned int u = __float_as_uint(f);
  u = (u + 0x7FFFu + ((u >> 16) & 1u)) >> 16;   // round-to-nearest-even
  return (unsigned short)u;
}
static __device__ __forceinline__ float sigmoidf_(float x) {
  return 1.0f / (1.0f + __expf(-x));
}

// ---------------------- pack f32 -> bf16 (with padding) ---------------------
// dst[r*dstCols + c] = (r<srcRows && c<srcCols) ? bf16(src[r*srcStride+c]) : 0
__global__ void k_pack_bf16(unsigned short* __restrict__ dst,
                            const float* __restrict__ src,
                            int dstRows, int dstCols,
                            int srcRows, int srcCols, int srcStride) {
  int i = blockIdx.x * blockDim.x + threadIdx.x;
  int total = dstRows * dstCols;
  if (i >= total) return;
  int r = i / dstCols, c = i % dstCols;
  float v = (r < srcRows && c < srcCols) ? src[(size_t)r * srcStride + c] : 0.0f;
  dst[i] = f2bf(v);
}

// ---------------------- patch extraction (im2row) -> bf16 -------------------
// dst[(ni*196+p)*768 + k], k = c*256 + py*16 + px
__global__ void k_im2row(unsigned short* __restrict__ dst,
                         const float* __restrict__ x) {
  int i = blockIdx.x * blockDim.x + threadIdx.x;
  if (i >= NTOK * 768) return;
  int row = i / 768, k = i % 768;
  int ni = row / 196, p = row % 196;
  int c = k / 256, rem = k % 256, py = rem / 16, px = rem % 16;
  int pi = p / 14, pj = p % 14;
  size_t idx = (((size_t)ni * 3 + c) * 224 + (pi * 16 + py)) * 224 + (pj * 16 + px);
  dst[i] = f2bf(x[idx]);
}

// ---------------------- bf16 WMMA GEMM --------------------------------------
// C[M,N](f32) = A[M,K](bf16,row-major) * B[N,K](bf16,row-major)^T (+ bias[N])
// Requires M%128==0, N%64==0, K%32==0. Block = 256 thr (8 wave32 waves).
// Wave (wm,wn) owns a 32x32 block = 2x2 WMMA tiles.
#define BM 128
#define BN 64
#define BK 32
#define LDA 40   // halves; row byte stride 80 (16B aligned)
#define LDB 40

__global__ __launch_bounds__(256)
void k_gemm_bf16(float* __restrict__ C,
                 const unsigned short* __restrict__ A,
                 const unsigned short* __restrict__ Bw,
                 const float* __restrict__ bias,
                 int M, int N, int K) {
  __shared__ __align__(16) unsigned short As[BM * LDA];
  __shared__ __align__(16) unsigned short Bs[BN * LDB];

  const int tid  = threadIdx.x;
  const int m0   = blockIdx.y * BM;
  const int n0   = blockIdx.x * BN;
  const int wave = tid >> 5, lane = tid & 31;
  const int wm   = wave & 3, wn = wave >> 2;       // 4 x 2 wave grid
  const int half = lane >> 4, lr = lane & 15;      // ISA 16-bit frag layout

  v8f acc[2][2] = {};

  const int ar = tid >> 1, akk = (tid & 1) * 16;   // A: 2 thr/row, 32B each
  const int br = tid >> 2, bkk = (tid & 3) * 8;    // B: 4 thr/row, 16B each
  const unsigned short* aG = A  + (size_t)(m0 + ar) * K + akk;
  const unsigned short* bG = Bw + (size_t)(n0 + br) * K + bkk;

  for (int k0 = 0; k0 < K; k0 += BK) {
    // cooperative global -> LDS staging
    {
      const uint4* g = (const uint4*)(aG + k0);
      uint4* s = (uint4*)(As + ar * LDA + akk);
      s[0] = g[0]; s[1] = g[1];
    }
    {
      const uint4* g = (const uint4*)(bG + k0);
      uint4* s = (uint4*)(Bs + br * LDB + bkk);
      s[0] = g[0];
    }
    if (k0 + BK < K) {                              // CDNA5 global_prefetch_b8
      __builtin_prefetch(aG + k0 + BK, 0, 1);
      __builtin_prefetch(bG + k0 + BK, 0, 1);
    }
    __syncthreads();

    // fragments per ISA 7.12.2: lane<16 -> K{0..7,16..23}, lane>=16 -> +8
    FragBf a[2], b[2];
#pragma unroll
    for (int i = 0; i < 2; ++i) {
      const uint4* p = (const uint4*)(As + (wm * 32 + i * 16 + lr) * LDA + half * 8);
      a[i].u[0] = p[0]; a[i].u[1] = p[2];
    }
#pragma unroll
    for (int j = 0; j < 2; ++j) {
      const uint4* p = (const uint4*)(Bs + (wn * 32 + j * 16 + lr) * LDB + half * 8);
      b[j].u[0] = p[0]; b[j].u[1] = p[2];
    }
#pragma unroll
    for (int i = 0; i < 2; ++i)
#pragma unroll
      for (int j = 0; j < 2; ++j)
        acc[i][j] = __builtin_amdgcn_wmma_f32_16x16x32_bf16(
            false, a[i].v, false, b[j].v, (short)0, acc[i][j], false, false);
    __syncthreads();
  }

  // C/D layout: lane<16 -> (M=r, N=lane); lane>=16 -> (M=r+8, N=lane-16)
#pragma unroll
  for (int i = 0; i < 2; ++i) {
#pragma unroll
    for (int j = 0; j < 2; ++j) {
      int mBase = m0 + wm * 32 + i * 16 + half * 8;
      int ncol  = n0 + wn * 32 + j * 16 + lr;
      float bv  = bias ? bias[ncol] : 0.0f;
#pragma unroll
      for (int r = 0; r < 8; ++r)
        C[(size_t)(mBase + r) * N + ncol] = acc[i][j][r] + bv;
    }
  }
}

// ---------------------- layernorm (one wave32 per row) ----------------------
__global__ void k_layernorm(float* __restrict__ out, const float* __restrict__ in,
                            const float* __restrict__ g, const float* __restrict__ b,
                            int rows) {
  int wave = threadIdx.x >> 5, lane = threadIdx.x & 31;
  int row = blockIdx.x * 8 + wave;
  if (row >= rows) return;
  const float* x = in + (size_t)row * DM;
  float v[12], s = 0.f, s2 = 0.f;
#pragma unroll
  for (int j = 0; j < 12; ++j) {
    v[j] = x[lane + j * 32];
    s += v[j]; s2 += v[j] * v[j];
  }
#pragma unroll
  for (int off = 16; off > 0; off >>= 1) {
    s  += __shfl_xor(s,  off, 32);
    s2 += __shfl_xor(s2, off, 32);
  }
  float mu = s * (1.0f / DM);
  float var = s2 * (1.0f / DM) - mu * mu;
  float rs = rsqrtf(var + 1e-5f);
  float* o = out + (size_t)row * DM;
#pragma unroll
  for (int j = 0; j < 12; ++j) {
    int e = lane + j * 32;
    o[e] = (v[j] - mu) * rs * g[e] + b[e];
  }
}

// ---------------------- depthwise conv (D_CONV=4) + SiLU --------------------
// xz: [NTOK,1536] (xi = cols 0..767); xc out: [NTOK,768]
__global__ void k_conv_silu(float* __restrict__ xc, const float* __restrict__ xz,
                            const float* __restrict__ cw, const float* __restrict__ cb) {
  int i = blockIdx.x * blockDim.x + threadIdx.x;
  if (i >= NTOK * DI) return;
  int tok = i / DI, e = i % DI;
  int b = tok / LSEQ, l = tok % LSEQ;
  float acc = cb[e];
#pragma unroll
  for (int k = 0; k < 4; ++k) {
    int ll = l - 3 + k;
    if (ll >= 0)
      acc += cw[e * 4 + k] * xz[(size_t)(b * LSEQ + ll) * 1536 + e];
  }
  xc[i] = acc * sigmoidf_(acc);
}

// ---------------------- dt bias + softplus ----------------------------------
__global__ void k_softplus_bias(float* __restrict__ dt, const float* __restrict__ dtb,
                                int n) {
  int i = blockIdx.x * blockDim.x + threadIdx.x;
  if (i >= n) return;
  float x = dt[i] + dtb[i % DI];
  dt[i] = (x > 20.0f) ? x : log1pf(__expf(x));
}

// ---------------------- selective scan --------------------------------------
// One thread per (batch, channel); B_t/C_t broadcast via LDS.
__global__ void k_scan(float* __restrict__ y, const float* __restrict__ dt,
                       const float* __restrict__ dbl, const float* __restrict__ xc,
                       const float* __restrict__ A_log, const float* __restrict__ Dp) {
  __shared__ float sBC[2 * DSTATE];
  int b = blockIdx.x / 6;
  int e = (blockIdx.x % 6) * 128 + threadIdx.x;
  float a[DSTATE], hst[DSTATE];
#pragma unroll
  for (int s = 0; s < DSTATE; ++s) {
    a[s] = -__expf(A_log[e * DSTATE + s]);
    hst[s] = 0.0f;
  }
  float dcoef = Dp[e];
  for (int l = 0; l < LSEQ; ++l) {
    int row = b * LSEQ + l;
    if (threadIdx.x < 2 * DSTATE)
      sBC[threadIdx.x] = dbl[(size_t)row * 64 + DTRANK + threadIdx.x];
    __syncthreads();
    float dtv = dt[(size_t)row * DI + e];
    float xv  = xc[(size_t)row * DI + e];
    float dx = dtv * xv;
    float acc = 0.0f;
#pragma unroll
    for (int s = 0; s < DSTATE; ++s) {
      hst[s] = __expf(dtv * a[s]) * hst[s] + dx * sBC[s];
      acc += hst[s] * sBC[DSTATE + s];
    }
    y[(size_t)row * DI + e] = acc + dcoef * xv;
    __syncthreads();
  }
}

// ---------------------- gating: y *= silu(z) --------------------------------
__global__ void k_gate(float* __restrict__ y, const float* __restrict__ xz, int n) {
  int i = blockIdx.x * blockDim.x + threadIdx.x;
  if (i >= n) return;
  int tok = i / DI, e = i % DI;
  float z = xz[(size_t)tok * 1536 + DI + e];
  y[i] *= z * sigmoidf_(z);
}

// ---------------------- residual add ----------------------------------------
__global__ void k_add(float* __restrict__ a, const float* __restrict__ b, int n) {
  int i = blockIdx.x * blockDim.x + threadIdx.x;
  if (i < n) a[i] += b[i];
}

// ---------------------- GELU (tanh approx) ----------------------------------
__global__ void k_gelu(float* __restrict__ a, int n) {
  int i = blockIdx.x * blockDim.x + threadIdx.x;
  if (i >= n) return;
  float x = a[i];
  float t = tanhf(0.7978845608028654f * (x + 0.044715f * x * x * x));
  a[i] = 0.5f * x * (1.0f + t);
}

// ---------------------- max-pool over patches + head ------------------------
__global__ void k_head(float* __restrict__ out, const float* __restrict__ h,
                       const float* __restrict__ w, const float* __restrict__ hb) {
  __shared__ float sp[DM];
  int nt = blockIdx.x;               // b*16 + t
  int e = threadIdx.x;
  int b = nt >> 4, t = nt & 15;
  const float* base = h + ((size_t)(b * LSEQ + t * 196)) * DM + e;
  float m = -3.4e38f;
  for (int p = 0; p < 196; ++p) m = fmaxf(m, base[(size_t)p * DM]);
  sp[e] = m;
  __syncthreads();
  if (e < 3) {
    float acc = hb[e];
    for (int k = 0; k < DM; ++k) acc += sp[k] * w[e * DM + k];
    out[nt * 3 + e] = acc;
  }
}

// ---------------------------------------------------------------------------

extern "C" void kernel_launch(void* const* d_in, const int* in_sizes, int n_in,
                              void* d_out, int out_size, void* d_ws, size_t ws_size,
                              hipStream_t stream) {
  (void)in_sizes; (void)n_in; (void)out_size; (void)ws_size;

  const float* x       = (const float*)d_in[0];
  const float* patch_w = (const float*)d_in[1];
  const float* patch_b = (const float*)d_in[2];
  const float* ln1_g   = (const float*)d_in[3];
  const float* ln1_b   = (const float*)d_in[4];
  const float* in_w    = (const float*)d_in[5];
  const float* conv_w  = (const float*)d_in[6];
  const float* conv_b  = (const float*)d_in[7];
  const float* xproj_w = (const float*)d_in[8];
  const float* dt_w    = (const float*)d_in[9];
  const float* dt_b    = (const float*)d_in[10];
  const float* A_log   = (const float*)d_in[11];
  const float* Dparam  = (const float*)d_in[12];
  const float* out_w   = (const float*)d_in[13];
  const float* ln2_g   = (const float*)d_in[14];
  const float* ln2_b   = (const float*)d_in[15];
  const float* fc1_w   = (const float*)d_in[16];
  const float* fc1_b   = (const float*)d_in[17];
  const float* fc2_w   = (const float*)d_in[18];
  const float* fc2_b   = (const float*)d_in[19];
  const float* head_w  = (const float*)d_in[20];
  const float* head_b  = (const float*)d_in[21];
  float* outp = (float*)d_out;

  // bump allocator over workspace
  char* wsb = (char*)d_ws;
  size_t off = 0;
  auto alloc = [&](size_t bytes) -> void* {
    void* p = wsb + off;
    off = (off + bytes + 255) & ~(size_t)255;
    return p;
  };

  unsigned short* wbf_patch = (unsigned short*)alloc((size_t)DM * 768 * 2);
  unsigned short *wbf_in[4], *wbf_xp[4], *wbf_dt[4], *wbf_out[4], *wbf_fc1[4], *wbf_fc2[4];
  for (int i = 0; i < 4; ++i) {
    wbf_in[i]  = (unsigned short*)alloc((size_t)1536 * 384 * 2);
    wbf_xp[i]  = (unsigned short*)alloc((size_t)64 * 768 * 2);
    wbf_dt[i]  = (unsigned short*)alloc((size_t)768 * 32 * 2);
    wbf_out[i] = (unsigned short*)alloc((size_t)384 * 768 * 2);
    wbf_fc1[i] = (unsigned short*)alloc((size_t)1536 * 384 * 2);
    wbf_fc2[i] = (unsigned short*)alloc((size_t)384 * 1536 * 2);
  }
  unsigned short* abf = (unsigned short*)alloc((size_t)NTOK * 1536 * 2);
  float* buf0 = (float*)alloc((size_t)NTOK * 1536 * 4);  // xz / mlp mid
  float* buf1 = (float*)alloc((size_t)NTOK * 768 * 4);   // xc
  float* buf2 = (float*)alloc((size_t)NTOK * 768 * 4);   // dt
  float* buf3 = (float*)alloc((size_t)NTOK * 768 * 4);   // y
  float* buf4 = (float*)alloc((size_t)NTOK * 384 * 4);   // layernorm out
  float* buf5 = (float*)alloc((size_t)NTOK * 384 * 4);   // block delta
  float* dblb = (float*)alloc((size_t)NTOK * 64 * 4);    // x_proj out (padded)
  float* hbuf = (float*)alloc((size_t)NTOK * 384 * 4);   // residual stream

  auto packW = [&](unsigned short* dst, const float* src, int dr, int dc,
                   int sr, int sc, int ss) {
    int n = dr * dc;
    k_pack_bf16<<<(n + 255) / 256, 256, 0, stream>>>(dst, src, dr, dc, sr, sc, ss);
  };
  auto gemm = [&](float* C, const unsigned short* A, const unsigned short* B,
                  const float* bias, int M, int N, int K) {
    dim3 grid(N / BN, M / BM);
    k_gemm_bf16<<<grid, 256, 0, stream>>>(C, A, B, bias, M, N, K);
  };

  // ---- one-time (per launch) weight packing to bf16 ----
  packW(wbf_patch, patch_w, 384, 768, 384, 768, 768);
  for (int i = 0; i < 4; ++i) {
    packW(wbf_in[i],  in_w    + (size_t)i * 1536 * 384, 1536, 384, 1536, 384, 384);
    packW(wbf_xp[i],  xproj_w + (size_t)i * 52 * 768,   64,   768, 52,   768, 768);
    packW(wbf_dt[i],  dt_w    + (size_t)i * 768 * 24,   768,  32,  768,  24,  24);
    packW(wbf_out[i], out_w   + (size_t)i * 384 * 768,  384,  768, 384,  768, 768);
    packW(wbf_fc1[i], fc1_w   + (size_t)i * 1536 * 384, 1536, 384, 1536, 384, 384);
    packW(wbf_fc2[i], fc2_w   + (size_t)i * 384 * 1536, 384,  1536, 384, 1536, 1536);
  }

  const int nTokD  = NTOK * 768;
  const int nTokDM = NTOK * 384;
  const int nTok2D = NTOK * 1536;

  // ---- patch embedding ----
  k_im2row<<<(nTokD + 255) / 256, 256, 0, stream>>>(abf, x);
  gemm(hbuf, abf, wbf_patch, patch_b, NTOK, 384, 768);

  // ---- 4 Mamba + MLP blocks ----
  for (int i = 0; i < 4; ++i) {
    // Mamba branch
    k_layernorm<<<NTOK / 8, 256, 0, stream>>>(buf4, hbuf, ln1_g + i * 384, ln1_b + i * 384, NTOK);
    packW(abf, buf4, NTOK, 384, NTOK, 384, 384);
    gemm(buf0, abf, wbf_in[i], nullptr, NTOK, 1536, 384);                  // xz
    k_conv_silu<<<(nTokD + 255) / 256, 256, 0, stream>>>(buf1, buf0,
                 conv_w + (size_t)i * 768 * 4, conv_b + i * 768);          // xc
    packW(abf, buf1, NTOK, 768, NTOK, 768, 768);
    gemm(dblb, abf, wbf_xp[i], nullptr, NTOK, 64, 768);                    // dbl (padded N)
    packW(abf, dblb, NTOK, 32, NTOK, 24, 64);                              // dt-rank slice (pad K)
    gemm(buf2, abf, wbf_dt[i], nullptr, NTOK, 768, 32);                    // dt raw
    k_softplus_bias<<<(nTokD + 255) / 256, 256, 0, stream>>>(buf2, dt_b + i * 768, nTokD);
    k_scan<<<12, 128, 0, stream>>>(buf3, buf2, dblb, buf1,
                                   A_log + (size_t)i * 768 * 14, Dparam + i * 768);
    k_gate<<<(nTokD + 255) / 256, 256, 0, stream>>>(buf3, buf0, nTokD);
    packW(abf, buf3, NTOK, 768, NTOK, 768, 768);
    gemm(buf5, abf, wbf_out[i], nullptr, NTOK, 384, 768);
    k_add<<<(nTokDM + 255) / 256, 256, 0, stream>>>(hbuf, buf5, nTokDM);

    // MLP branch
    k_layernorm<<<NTOK / 8, 256, 0, stream>>>(buf4, hbuf, ln2_g + i * 384, ln2_b + i * 384, NTOK);
    packW(abf, buf4, NTOK, 384, NTOK, 384, 384);
    gemm(buf0, abf, wbf_fc1[i], fc1_b + i * 1536, NTOK, 1536, 384);
    k_gelu<<<(nTok2D + 255) / 256, 256, 0, stream>>>(buf0, nTok2D);
    packW(abf, buf0, NTOK, 1536, NTOK, 1536, 1536);
    gemm(buf5, abf, wbf_fc2[i], fc2_b + i * 384, NTOK, 384, 1536);
    k_add<<<(nTokDM + 255) / 256, 256, 0, stream>>>(hbuf, buf5, nTokDM);
  }

  // ---- max-pool over patches + classifier head ----
  k_head<<<32, 384, 0, stream>>>(outp, hbuf, head_w, head_b);
}
